// VariationalAutoEncoder_50560355008707
// MI455X (gfx1250) — matrix operand order repaired and checked by reading, not compiled, not run on hardware
//
#include <hip/hip_runtime.h>

typedef __attribute__((ext_vector_type(16))) _Float16 v16h;
typedef __attribute__((ext_vector_type(8)))  _Float16 v8h;
typedef __attribute__((ext_vector_type(8)))  float    v8f;

#define N_TOT   98304
#define EDGES   1572864
#define BATCHG  2048
#define DIN     128
#define HID     512
#define LATD    64
#define NMAXN   96
#define MPAIRS  4560
#define BN_EPS  1e-5f

// ---------------------------------------------------------------------------
// small float4 helpers (avoid relying on vector operator overloads)
// ---------------------------------------------------------------------------
__device__ __forceinline__ float4 f4add(float4 a, float4 b) {
    return make_float4(a.x + b.x, a.y + b.y, a.z + b.z, a.w + b.w);
}
__device__ __forceinline__ float4 f4fma(float4 a, float4 s, float4 t) {
    return make_float4(fmaf(a.x, s.x, t.x), fmaf(a.y, s.y, t.y),
                       fmaf(a.z, s.z, t.z), fmaf(a.w, s.w, t.w));
}

// ---------------------------------------------------------------------------
// Weight pre-pack: f32 [K,N] row-major -> f16 fragment-native layout
//   P[((k/32)*Np + n)*32 + (k%32)], zero-padded to Kp (mult 32) and Np.
// A lane's 16-half B fragment is then contiguous: two global_load_b128.
// ---------------------------------------------------------------------------
__global__ __launch_bounds__(256)
void pack_w(const float* __restrict__ W, _Float16* __restrict__ P,
            int K, int N, int Kp, int Np)
{
    long idx = (long)blockIdx.x * 256 + threadIdx.x;
    long total = (long)Kp * Np;
    if (idx >= total) return;
    int  kin  = (int)(idx & 31);
    long t    = idx >> 5;
    int  n    = (int)(t % Np);
    int  kblk = (int)(t / Np);
    int  k    = kblk * 32 + kin;
    float v = (k < K && n < N) ? W[(long)k * N + n] : 0.0f;
    P[idx] = (_Float16)v;
}

// ---------------------------------------------------------------------------
// WMMA GEMM (guard-free fast path): C[M,N] = epi( A'[M,K] @ W[K,N] + bias )
//   AMODE 0: A' = A
//   AMODE 1: A' = A + A2                       (GIN: h + aggregate)
//   AMODE 2: A' = A*scale[k] + shift[k]        (fused BatchNorm, prefolded)
//   EPI   0: none   1: leaky_relu(0.2)   2: relu
//   NG: guard bias/store columns against ragged N (loads are pad-safe)
// Requirements: M%128==0, K%32==0, lda%4==0; B packed with Np >= 64*gridDim.x.
// Block: 256 threads = 8 waves; wave w -> rows [w*16,w*16+16) x 4 col-tiles.
// ---------------------------------------------------------------------------
template<int AMODE, int EPI, bool NG>
__global__ __launch_bounds__(256)
void gemm_f16(const float* __restrict__ A, const float* __restrict__ A2,
              const float* __restrict__ scale, const float* __restrict__ shift,
              const _Float16* __restrict__ P, const float* __restrict__ bias,
              float* __restrict__ C, int M, int N, int K, int lda, int Np)
{
    const int tid  = threadIdx.x;
    const int lane = tid & 31;
    const int wid  = tid >> 5;
    const int hi   = (lane >> 4) & 1;
    const int l16  = lane & 15;
    const int rowA = blockIdx.y * 128 + wid * 16 + l16;   // A-fragment row
    const int colBase = blockIdx.x * 64;
    const int klo = hi ? 16 : 0;     // contiguous k = 16*hi + s, halves 0..7
    const int khi = hi ? 24 : 8;     // halves 8..15

    v8f acc[4];
    #pragma unroll
    for (int nt = 0; nt < 4; ++nt)
        #pragma unroll
        for (int i = 0; i < 8; ++i) acc[nt][i] = 0.0f;

    for (int k0 = 0; k0 < K; k0 += 32) {
        // ---- A fragment: 16 f32 per lane, four aligned float4 loads
        const float* ap = A + (long)rowA * lda + k0;
        float4 x0 = *(const float4*)(ap + klo);
        float4 x1 = *(const float4*)(ap + klo + 4);
        float4 x2 = *(const float4*)(ap + khi);
        float4 x3 = *(const float4*)(ap + khi + 4);
        if (AMODE == 1) {
            const float* ap2 = A2 + (long)rowA * lda + k0;
            x0 = f4add(x0, *(const float4*)(ap2 + klo));
            x1 = f4add(x1, *(const float4*)(ap2 + klo + 4));
            x2 = f4add(x2, *(const float4*)(ap2 + khi));
            x3 = f4add(x3, *(const float4*)(ap2 + khi + 4));
        } else if (AMODE == 2) {
            x0 = f4fma(x0, *(const float4*)(scale + k0 + klo),
                           *(const float4*)(shift + k0 + klo));
            x1 = f4fma(x1, *(const float4*)(scale + k0 + klo + 4),
                           *(const float4*)(shift + k0 + klo + 4));
            x2 = f4fma(x2, *(const float4*)(scale + k0 + khi),
                           *(const float4*)(shift + k0 + khi));
            x3 = f4fma(x3, *(const float4*)(scale + k0 + khi + 4),
                           *(const float4*)(shift + k0 + khi + 4));
        }
        v16h af;
        af[0]  = (_Float16)x0.x; af[1]  = (_Float16)x0.y;
        af[2]  = (_Float16)x0.z; af[3]  = (_Float16)x0.w;
        af[4]  = (_Float16)x1.x; af[5]  = (_Float16)x1.y;
        af[6]  = (_Float16)x1.z; af[7]  = (_Float16)x1.w;
        af[8]  = (_Float16)x2.x; af[9]  = (_Float16)x2.y;
        af[10] = (_Float16)x2.z; af[11] = (_Float16)x2.w;
        af[12] = (_Float16)x3.x; af[13] = (_Float16)x3.y;
        af[14] = (_Float16)x3.z; af[15] = (_Float16)x3.w;

        // ---- B fragments: packed layout -> 16 contiguous halves per lane
        const _Float16* pk = P + ((long)(k0 >> 5) * Np) * 32 + (hi ? 16 : 0);
        #pragma unroll
        for (int nt = 0; nt < 4; ++nt) {
            const int col = colBase + nt * 16 + l16;
            const _Float16* q = pk + (long)col * 32;
            v8h b0 = *(const v8h*)q;
            v8h b1 = *(const v8h*)(q + 8);
            v16h bf;
            #pragma unroll
            for (int i = 0; i < 8; ++i) { bf[i] = b0[i]; bf[i + 8] = b1[i]; }
            acc[nt] = __builtin_amdgcn_wmma_f32_16x16x32_f16(
                false, af, false, bf, (short)0, acc[nt], false, false);
        }
    }

    // ---- epilogue: C/D layout — VGPR r: M = r + 8*hi, N = lane%16
    #pragma unroll
    for (int nt = 0; nt < 4; ++nt) {
        const int gc = colBase + nt * 16 + l16;
        if (NG && gc >= N) continue;
        const float bv = bias[gc];
        #pragma unroll
        for (int r = 0; r < 8; ++r) {
            const int gr = blockIdx.y * 128 + wid * 16 + (hi ? 8 : 0) + r;
            float v = acc[nt][r] + bv;
            if (EPI == 1)      v = v > 0.0f ? v : 0.2f * v;
            else if (EPI == 2) v = v > 0.0f ? v : 0.0f;
            C[(long)gr * N + gc] = v;
        }
    }
}

// ---------------------------------------------------------------------------
__global__ void zero_kernel(float* __restrict__ p, long n)
{
    long i = (long)blockIdx.x * blockDim.x + threadIdx.x;
    long stride = (long)gridDim.x * blockDim.x;
    for (; i < n; i += stride) p[i] = 0.0f;
}

// agg[dst] += h[src] over edges (float4 per thread)
__global__ __launch_bounds__(256)
void scatter_kernel(const float* __restrict__ h, const int* __restrict__ src,
                    const int* __restrict__ dst, float* __restrict__ agg, int D4)
{
    long idx = (long)blockIdx.x * 256 + threadIdx.x;
    long total = (long)EDGES * D4;
    if (idx >= total) return;
    int e  = (int)(idx / D4);
    int c4 = (int)(idx % D4) << 2;
    int D  = D4 << 2;
    int s = src[e], d = dst[e];
    const float4 v = *(const float4*)&h[(long)s * D + c4];
    float* p = &agg[(long)d * D + c4];
    atomicAdd(p + 0, v.x); atomicAdd(p + 1, v.y);
    atomicAdd(p + 2, v.z); atomicAdd(p + 3, v.w);
}

// per-column BatchNorm stats, prefolded into affine: scale = g*rstd,
// shift = beta - mean*scale  (so BN(x) = x*scale + shift)
__global__ __launch_bounds__(256)
void colstats(const float* __restrict__ X, int M, int C,
              const float* __restrict__ g, const float* __restrict__ bt,
              float* __restrict__ scale, float* __restrict__ shift)
{
    __shared__ float s1[256], s2[256];
    int c = blockIdx.x;
    float a = 0.0f, b = 0.0f;
    for (int r = threadIdx.x; r < M; r += 256) {
        float v = X[(long)r * C + c];
        a += v; b += v * v;
    }
    s1[threadIdx.x] = a; s2[threadIdx.x] = b;
    __syncthreads();
    for (int s = 128; s > 0; s >>= 1) {
        if (threadIdx.x < s) {
            s1[threadIdx.x] += s1[threadIdx.x + s];
            s2[threadIdx.x] += s2[threadIdx.x + s];
        }
        __syncthreads();
    }
    if (threadIdx.x == 0) {
        float m  = s1[0] / (float)M;
        float v  = s2[0] / (float)M - m * m;
        float sc = g[c] * rsqrtf(v + BN_EPS);
        scale[c] = sc;
        shift[c] = bt[c] - m * sc;
    }
}

// att[i] = sigmoid(dot(h[i,:512], w) + b)  — one wave32 per node
__global__ __launch_bounds__(256)
void att_kernel(const float* __restrict__ h, const float* __restrict__ w,
                const float* __restrict__ b, float* __restrict__ att, int n)
{
    int wid = threadIdx.x >> 5, lane = threadIdx.x & 31;
    int node = blockIdx.x * 8 + wid;
    if (node >= n) return;
    float s = 0.0f;
    #pragma unroll
    for (int t = 0; t < 16; ++t) {
        int c = t * 32 + lane;
        s += h[(long)node * HID + c] * w[c];
    }
    #pragma unroll
    for (int off = 16; off > 0; off >>= 1) s += __shfl_xor(s, off, 32);
    if (lane == 0) att[node] = 1.0f / (1.0f + __expf(-(s + b[0])));
}

// pooled[batch[i]] += h[i] * att[i]
__global__ __launch_bounds__(256)
void pool_kernel(const float* __restrict__ h, const float* __restrict__ att,
                 const int* __restrict__ batch, float* __restrict__ pooled, int n)
{
    long idx = (long)blockIdx.x * 256 + threadIdx.x;
    long total = (long)n * (HID / 4);
    if (idx >= total) return;
    int node = (int)(idx >> 7);
    int c4   = (int)(idx & 127) << 2;
    const float4 v = *(const float4*)&h[(long)node * HID + c4];
    float a = att[node];
    int gidx = batch[node];
    float* p = &pooled[(long)gidx * HID + c4];
    atomicAdd(p + 0, v.x * a); atomicAdd(p + 1, v.y * a);
    atomicAdd(p + 2, v.z * a); atomicAdd(p + 3, v.w * a);
}

// z = mu + eps*exp(0.5*logvar); d0 = concat(z, stats) zero-padded to 96 cols
__global__ __launch_bounds__(256)
void zconcat_kernel(const float* __restrict__ mu, const float* __restrict__ lv,
                    const float* __restrict__ eps, const float* __restrict__ stats,
                    float* __restrict__ d0)
{
    int idx = blockIdx.x * 256 + threadIdx.x;
    if (idx >= BATCHG * 96) return;
    int b = idx / 96, j = idx % 96;
    float v = 0.0f;
    if (j < LATD)      v = mu[b * LATD + j] + eps[b * LATD + j] * __expf(0.5f * lv[b * LATD + j]);
    else if (j < 71)   v = stats[b * 7 + (j - LATD)];
    d0[b * 96 + j] = v;
}

// hard gumbel-softmax ST -> symmetric masked adjacency, one thread per cell
__global__ __launch_bounds__(256)
void adj_kernel(const float* __restrict__ logits, const float* __restrict__ gumbel,
                const float* __restrict__ stats, float* __restrict__ adj)
{
    long idx = (long)blockIdx.x * 256 + threadIdx.x;
    if (idx >= (long)BATCHG * NMAXN * NMAXN) return;
    int b   = (int)(idx / (NMAXN * NMAXN));
    int rem = (int)(idx % (NMAXN * NMAXN));
    int i = rem / NMAXN, j = rem % NMAXN;
    float out = 0.0f;
    if (i != j) {
        int lo = min(i, j), hi2 = max(i, j);
        int m = lo * (NMAXN - 1) - (lo * (lo - 1)) / 2 + (hi2 - lo - 1);
        long base = (long)b * 2 * MPAIRS + 2 * m;
        float a0 = logits[base]     + gumbel[base];
        float a1 = logits[base + 1] + gumbel[base + 1];
        float mx = fmaxf(a0, a1);
        float e0 = __expf(a0 - mx), e1 = __expf(a1 - mx);
        float s0 = e0 / (e0 + e1);
        float hard = (a0 >= a1) ? 1.0f : 0.0f;
        float y = (hard - s0) + s0;   // straight-through forward value
        int nb = (int)stats[b * 7];
        if (i < nb && j < nb) out = y;
    }
    adj[idx] = out;
}

// ---------------------------------------------------------------------------
static inline int cdiv(long a, long b) { return (int)((a + b - 1) / b); }

extern "C" void kernel_launch(void* const* d_in, const int* in_sizes, int n_in,
                              void* d_out, int out_size, void* d_ws, size_t ws_size,
                              hipStream_t stream)
{
    (void)in_sizes; (void)n_in; (void)out_size; (void)ws_size;

    const float* x      = (const float*)d_in[0];
    const int*   eidx   = (const int*)d_in[1];
    const int*   src    = eidx;
    const int*   dst    = eidx + EDGES;
    const int*   batch  = (const int*)d_in[2];
    const float* stats  = (const float*)d_in[3];
    const float* eps_z  = (const float*)d_in[4];
    const float* gumbel = (const float*)d_in[5];
    // params flattened in reference insertion order:
    // convs[l]: l1.w, l1.b, bn_g, bn_b, l2.w, l2.b   (indices 6 + 6*l .. +5)
    const float* att_w = (const float*)d_in[30];
    const float* att_b = (const float*)d_in[31];
    const float* bng_g = (const float*)d_in[32];
    const float* bng_b = (const float*)d_in[33];
    const float* flt_w = (const float*)d_in[34];
    const float* flt_b = (const float*)d_in[35];
    const float* fmu_w = (const float*)d_in[36];
    const float* fmu_b = (const float*)d_in[37];
    const float* flv_w = (const float*)d_in[38];
    const float* flv_b = (const float*)d_in[39];
    const float* dc0_w = (const float*)d_in[40];
    const float* dc0_b = (const float*)d_in[41];
    const float* dc1_w = (const float*)d_in[42];
    const float* dc1_b = (const float*)d_in[43];
    const float* dc2_w = (const float*)d_in[44];
    const float* dc2_b = (const float*)d_in[45];

    // workspace carve-up (keep every block 256B-aligned)
    float* ws = (float*)d_ws;
    size_t o = 0;
    auto alloc = [&](size_t nfloat) {
        float* p = ws + o;
        o += (nfloat + 63) & ~(size_t)63;
        return p;
    };
    float* agg    = alloc((size_t)N_TOT * HID);
    float* X1     = alloc((size_t)N_TOT * HID);
    float* hbuf   = alloc((size_t)N_TOT * HID);
    float* attb   = alloc((size_t)N_TOT);
    float* pooled = alloc((size_t)BATCHG * HID);
    float* xg     = alloc((size_t)BATCHG * LATD);
    float* d0     = alloc((size_t)BATCHG * 96);
    float* d1     = alloc((size_t)BATCHG * 512);
    float* d2     = alloc((size_t)BATCHG * 1024);
    float* logits = alloc((size_t)BATCHG * 2 * MPAIRS);
    float* scaleb = alloc(1024);
    float* shiftb = alloc(1024);
    auto allocH = [&](size_t nhalf) { return (_Float16*)alloc((nhalf + 1) / 2); };
    _Float16* pw_l1[4]; _Float16* pw_l2[4];
    pw_l1[0] = allocH((size_t)DIN * HID);
    for (int l = 1; l < 4; ++l) pw_l1[l] = allocH((size_t)HID * HID);
    for (int l = 0; l < 4; ++l) pw_l2[l] = allocH((size_t)HID * HID);
    _Float16* pw_lat = allocH((size_t)HID * LATD);
    _Float16* pw_mu  = allocH((size_t)LATD * LATD);
    _Float16* pw_lv  = allocH((size_t)LATD * LATD);
    _Float16* pw_d0  = allocH((size_t)96 * 512);
    _Float16* pw_d1  = allocH((size_t)512 * 1024);
    const int NP2 = 9152;                 // 9120 padded to 64
    _Float16* pw_d2  = allocH((size_t)1024 * NP2);

    // outputs: (adj, mu, logvar) flat
    float* adj = (float*)d_out;
    float* mu  = adj + (size_t)BATCHG * NMAXN * NMAXN;
    float* lv  = mu + (size_t)BATCHG * LATD;

    const dim3 blk(256);
    auto packW = [&](const float* W, _Float16* P, int K, int N, int Kp, int Np) {
        pack_w<<<cdiv((long)Kp * Np, 256), blk, 0, stream>>>(W, P, K, N, Kp, Np);
    };

    // ----- pack all weights to f16 fragment-native layout -----
    for (int l = 0; l < 4; ++l) {
        int D = (l == 0) ? DIN : HID;
        packW((const float*)d_in[6 + 6 * l + 0], pw_l1[l], D, HID, D, HID);
        packW((const float*)d_in[6 + 6 * l + 4], pw_l2[l], HID, HID, HID, HID);
    }
    packW(flt_w, pw_lat, HID, LATD, HID, LATD);
    packW(fmu_w, pw_mu, LATD, LATD, LATD, LATD);
    packW(flv_w, pw_lv, LATD, LATD, LATD, LATD);
    packW(dc0_w, pw_d0, 71, 512, 96, 512);
    packW(dc1_w, pw_d1, 512, 1024, 512, 1024);
    packW(dc2_w, pw_d2, 1024, 2 * MPAIRS, 1024, NP2);

    // ----- GIN encoder: 4 layers -----
    const float* hcur = x;
    for (int l = 0; l < 4; ++l) {
        int D = (l == 0) ? DIN : HID;
        const float* b1 = (const float*)d_in[6 + 6 * l + 1];
        const float* bg = (const float*)d_in[6 + 6 * l + 2];
        const float* bb = (const float*)d_in[6 + 6 * l + 3];
        const float* b2 = (const float*)d_in[6 + 6 * l + 5];

        zero_kernel<<<4096, blk, 0, stream>>>(agg, (long)N_TOT * D);
        scatter_kernel<<<cdiv((long)EDGES * (D / 4), 256), blk, 0, stream>>>(
            hcur, src, dst, agg, D / 4);

        // X1 = leaky( (h+agg) @ W1 + b1 )
        gemm_f16<1, 1, false><<<dim3(HID / 64, N_TOT / 128), blk, 0, stream>>>(
            hcur, agg, nullptr, nullptr, pw_l1[l], b1, X1, N_TOT, HID, D, D, HID);
        // BN stats folded to affine
        colstats<<<HID, blk, 0, stream>>>(X1, N_TOT, HID, bg, bb, scaleb, shiftb);
        // h = leaky( BN(X1) @ W2 + b2 )
        gemm_f16<2, 1, false><<<dim3(HID / 64, N_TOT / 128), blk, 0, stream>>>(
            X1, nullptr, scaleb, shiftb, pw_l2[l], b2, hbuf, N_TOT, HID, HID, HID, HID);
        hcur = hbuf;
    }

    // ----- attention pooling -----
    att_kernel<<<cdiv(N_TOT, 8), blk, 0, stream>>>(hbuf, att_w, att_b, attb, N_TOT);
    zero_kernel<<<1024, blk, 0, stream>>>(pooled, (long)BATCHG * HID);
    pool_kernel<<<cdiv((long)N_TOT * (HID / 4), 256), blk, 0, stream>>>(
        hbuf, attb, batch, pooled, N_TOT);
    colstats<<<HID, blk, 0, stream>>>(pooled, BATCHG, HID, bng_g, bng_b, scaleb, shiftb);

    // x_g = BN(pooled) @ W_lat + b_lat
    gemm_f16<2, 0, false><<<dim3(LATD / 64, BATCHG / 128), blk, 0, stream>>>(
        pooled, nullptr, scaleb, shiftb, pw_lat, flt_b, xg, BATCHG, LATD, HID, HID, LATD);
    // mu / logvar straight into d_out
    gemm_f16<0, 0, false><<<dim3(LATD / 64, BATCHG / 128), blk, 0, stream>>>(
        xg, nullptr, nullptr, nullptr, pw_mu, fmu_b, mu, BATCHG, LATD, LATD, LATD, LATD);
    gemm_f16<0, 0, false><<<dim3(LATD / 64, BATCHG / 128), blk, 0, stream>>>(
        xg, nullptr, nullptr, nullptr, pw_lv, flv_b, lv, BATCHG, LATD, LATD, LATD, LATD);

    // ----- decoder -----
    zconcat_kernel<<<cdiv(BATCHG * 96, 256), blk, 0, stream>>>(mu, lv, eps_z, stats, d0);
    gemm_f16<0, 2, false><<<dim3(512 / 64, BATCHG / 128), blk, 0, stream>>>(
        d0, nullptr, nullptr, nullptr, pw_d0, dc0_b, d1, BATCHG, 512, 96, 96, 512);
    gemm_f16<0, 2, false><<<dim3(1024 / 64, BATCHG / 128), blk, 0, stream>>>(
        d1, nullptr, nullptr, nullptr, pw_d1, dc1_b, d2, BATCHG, 1024, 512, 512, 1024);
    gemm_f16<0, 0, true><<<dim3(NP2 / 64, BATCHG / 128), blk, 0, stream>>>(
        d2, nullptr, nullptr, nullptr, pw_d2, dc2_b, logits, BATCHG, 2 * MPAIRS, 1024, 1024, NP2);

    // ----- adjacency assembly -----
    adj_kernel<<<cdiv((long)BATCHG * NMAXN * NMAXN, 256), blk, 0, stream>>>(
        logits, gumbel, stats, adj);
}
